// LocalAttention_25984552141263
// MI455X (gfx1250) — compile-verified, matrix operand order checked by reading, not run on hardware
//
#include <hip/hip_runtime.h>

// ---------------- problem constants ----------------
#define BATCH   4
#define NSEQ    4096
#define DIM_    1024
#define HEADS   16
#define HD      64
#define WIN     256
#define EXT_    128
#define GRP     16          // NSEQ / WIN
#define KW      512         // WIN + 2*EXT
#define SCALE_  0.125f      // HD^-0.5

typedef __attribute__((ext_vector_type(16))) __bf16 v16bf;
typedef __attribute__((ext_vector_type(8)))  __bf16 v8bf;
typedef __attribute__((ext_vector_type(8)))  float  v8f;

// ---------------- helpers ----------------
__device__ inline __bf16 f2bf(float f) {
    union { float f; unsigned u; } c; c.f = f;
    unsigned r = c.u + 0x7FFFu + ((c.u >> 16) & 1u);   // RNE
    unsigned short h = (unsigned short)(r >> 16);
    union { unsigned short s; __bf16 b; } o; o.s = h;
    return o.b;
}

// Load one 16x32 bf16 fragment in WMMA A/B layout from a row-major matrix.
// lane 0-15 : logical row = row0+lane,  K chunks {c0..c0+7 , c0+16..c0+23}
// lane16-31 : logical row = row0+lane-16, K chunks {c0+8.. , c0+24..}
__device__ inline v16bf load_frag(const __bf16* base, int row0, int col0, int ld) {
    int lane = threadIdx.x & 31;
    int half = lane >> 4;
    int r    = lane & 15;
    const __bf16* p = base + (size_t)(row0 + r) * ld + col0 + half * 8;
    v8bf lo = *(const v8bf*)(p);
    v8bf hi = *(const v8bf*)(p + 16);
    v16bf f;
#pragma unroll
    for (int i = 0; i < 8; ++i) { f[i] = lo[i]; f[i + 8] = hi[i]; }
    return f;
}

__device__ inline float red_max16(float v) {
#pragma unroll
    for (int off = 1; off < 16; off <<= 1) v = fmaxf(v, __shfl_xor(v, off, 32));
    return v;
}
__device__ inline float red_sum16(float v) {
#pragma unroll
    for (int off = 1; off < 16; off <<= 1) v += __shfl_xor(v, off, 32);
    return v;
}

#define WMMA_BF16(a, b, c) \
    __builtin_amdgcn_wmma_f32_16x16x32_bf16(false, (a), false, (b), (short)0, (c), false, false)

// ---------------- kernel 1: fp32 -> bf16 ----------------
__global__ __launch_bounds__(256) void cvt_bf16_kernel(const float* __restrict__ in,
                                                       __bf16* __restrict__ out, int n) {
    for (int i = blockIdx.x * 256 + threadIdx.x; i < n; i += gridDim.x * 256)
        out[i] = f2bf(in[i]);
}

// ---------------- kernel 2: QKV GEMM (M=16384, N=3072, K=1024) ----------------
// C = A * W^T + bias ; scatter to Q (b,h,n,d), K (b,h,n,d), Vt (b,h,d,n), bf16
__global__ __launch_bounds__(256) void qkv_gemm_kernel(const __bf16* __restrict__ A,
                                                       const __bf16* __restrict__ Wm,
                                                       const float*  __restrict__ bias,
                                                       __bf16* __restrict__ Qo,
                                                       __bf16* __restrict__ Ko,
                                                       __bf16* __restrict__ Vto) {
    const int mb = blockIdx.x * 128;
    const int nb = blockIdx.y * 128;
    const int wave = threadIdx.x >> 5;
    const int wm = (wave & 3) * 32;          // wave's M offset in block tile
    const int wn = (wave >> 2) * 64;         // wave's N offset in block tile
    const int lane = threadIdx.x & 31, half = lane >> 4, r = lane & 15;

    v8f acc[2][4];
#pragma unroll
    for (int i = 0; i < 2; ++i)
#pragma unroll
        for (int j = 0; j < 4; ++j) acc[i][j] = (v8f){};

    for (int k0 = 0; k0 < DIM_; k0 += 32) {
        v16bf a0 = load_frag(A, mb + wm,      k0, DIM_);
        v16bf a1 = load_frag(A, mb + wm + 16, k0, DIM_);
        v16bf b0 = load_frag(Wm, nb + wn,      k0, DIM_);
        v16bf b1 = load_frag(Wm, nb + wn + 16, k0, DIM_);
        v16bf b2 = load_frag(Wm, nb + wn + 32, k0, DIM_);
        v16bf b3 = load_frag(Wm, nb + wn + 48, k0, DIM_);
        acc[0][0] = WMMA_BF16(a0, b0, acc[0][0]);
        acc[0][1] = WMMA_BF16(a0, b1, acc[0][1]);
        acc[0][2] = WMMA_BF16(a0, b2, acc[0][2]);
        acc[0][3] = WMMA_BF16(a0, b3, acc[0][3]);
        acc[1][0] = WMMA_BF16(a1, b0, acc[1][0]);
        acc[1][1] = WMMA_BF16(a1, b1, acc[1][1]);
        acc[1][2] = WMMA_BF16(a1, b2, acc[1][2]);
        acc[1][3] = WMMA_BF16(a1, b3, acc[1][3]);
    }

#pragma unroll
    for (int i = 0; i < 2; ++i) {
#pragma unroll
        for (int j = 0; j < 4; ++j) {
            const int col = nb + wn + j * 16 + r;          // 0..3071
            const float bv = bias[col];
            const int which = col >> 10;                   // 0=q 1=k 2=v
            const int cc = col & (DIM_ - 1);
            const int h = cc >> 6, d = cc & (HD - 1);
#pragma unroll
            for (int jj = 0; jj < 8; ++jj) {
                const int row = mb + wm + i * 16 + jj + half * 8;   // 0..16383
                const int bI = row >> 12, n = row & (NSEQ - 1);
                const float val = acc[i][j][jj] + bv;
                const size_t hbase = (size_t)(bI * HEADS + h);
                if (which == 0)
                    Qo[(hbase * NSEQ + n) * HD + d] = f2bf(val);
                else if (which == 1)
                    Ko[(hbase * NSEQ + n) * HD + d] = f2bf(val);
                else
                    Vto[(hbase * HD + d) * NSEQ + n] = f2bf(val);
            }
        }
    }
}

// ---------------- kernel 3: flash local attention ----------------
// one block per (b,h,g); 8 waves, each owns 2 query tiles of 16 rows.
__global__ __launch_bounds__(256) void attn_kernel(const __bf16* __restrict__ Q,
                                                   const __bf16* __restrict__ K,
                                                   const __bf16* __restrict__ Vt,
                                                   __bf16* __restrict__ attnb) {
    __shared__ __bf16 plds[8][16 * 32];      // per-wave P strip (16 rows x 32 keys)

    const int bid = blockIdx.x;
    const int g = bid & 15, h = (bid >> 4) & 15, b = bid >> 8;
    const int wave = threadIdx.x >> 5;
    const int lane = threadIdx.x & 31, half = lane >> 4, r = lane & 15;

    const __bf16* Qbh  = Q  + ((size_t)(b * HEADS + h) * NSEQ) * HD;
    const __bf16* Kbh  = K  + ((size_t)(b * HEADS + h) * NSEQ) * HD;
    const __bf16* Vtbh = Vt + ((size_t)(b * HEADS + h) * HD) * NSEQ;

    // Q fragments: 2 q-tiles x 2 hd-chunks
    v16bf qa[2][2];
#pragma unroll
    for (int qt = 0; qt < 2; ++qt)
#pragma unroll
        for (int kc = 0; kc < 2; ++kc)
            qa[qt][kc] = load_frag(Qbh, g * WIN + wave * 32 + qt * 16, kc * 32, HD);

    float mrun[2][8], lrun[2][8];
    v8f O[2][4];
#pragma unroll
    for (int qt = 0; qt < 2; ++qt) {
#pragma unroll
        for (int j = 0; j < 8; ++j) { mrun[qt][j] = -1e30f; lrun[qt][j] = 0.f; }
#pragma unroll
        for (int t = 0; t < 4; ++t) O[qt][t] = (v8f){};
    }

    // pad regions align to 32-key chunks -> just clamp the loop
    const int kstart = (g == 0) ? 128 : 0;
    const int kend   = (g == GRP - 1) ? (KW - 128) : KW;

    for (int kw = kstart; kw < kend; kw += 32) {
        const int pos0 = g * WIN - EXT_ + kw;            // global key index of chunk
        // K B-fragments: 2 key groups x 2 hd-chunks
        v16bf kb[2][2];
        kb[0][0] = load_frag(Kbh, pos0,      0,  HD);
        kb[0][1] = load_frag(Kbh, pos0,      32, HD);
        kb[1][0] = load_frag(Kbh, pos0 + 16, 0,  HD);
        kb[1][1] = load_frag(Kbh, pos0 + 16, 32, HD);
        // V B-fragments: 4 hd tiles, k-dim = 32 keys (contiguous in Vt rows)
        v16bf vb[4];
#pragma unroll
        for (int t = 0; t < 4; ++t) vb[t] = load_frag(Vtbh, t * 16, pos0, NSEQ);

        for (int qt = 0; qt < 2; ++qt) {
            v8f s0 = (v8f){}, s1 = (v8f){};
            s0 = WMMA_BF16(qa[qt][0], kb[0][0], s0);
            s0 = WMMA_BF16(qa[qt][1], kb[0][1], s0);
            s1 = WMMA_BF16(qa[qt][0], kb[1][0], s1);
            s1 = WMMA_BF16(qa[qt][1], kb[1][1], s1);

#pragma unroll
            for (int j = 0; j < 8; ++j) {
                float a = s0[j] * SCALE_;
                float c = s1[j] * SCALE_;
                float rmax = red_max16(fmaxf(a, c));
                float mnew = fmaxf(mrun[qt][j], rmax);
                float corr = __expf(mrun[qt][j] - mnew);
                float p0 = __expf(a - mnew);
                float p1 = __expf(c - mnew);
                float rsum = red_sum16(p0 + p1);
                lrun[qt][j] = lrun[qt][j] * corr + rsum;
                mrun[qt][j] = mnew;
#pragma unroll
                for (int t = 0; t < 4; ++t) O[qt][t][j] *= corr;
                const int prow = j + half * 8;
                plds[wave][prow * 32 + r]      = f2bf(p0);
                plds[wave][prow * 32 + 16 + r] = f2bf(p1);
            }
            // LDS RAW across lanes within the wave: wait for ds stores
            asm volatile("s_wait_dscnt 0" ::: "memory");

            // reload P in A-fragment layout (two 16B LDS reads)
            const __bf16* pp = &plds[wave][r * 32 + half * 8];
            v8bf lo = *(const v8bf*)(pp);
            v8bf hi = *(const v8bf*)(pp + 16);
            v16bf pa;
#pragma unroll
            for (int i = 0; i < 8; ++i) { pa[i] = lo[i]; pa[i + 8] = hi[i]; }

#pragma unroll
            for (int t = 0; t < 4; ++t) O[qt][t] = WMMA_BF16(pa, vb[t], O[qt][t]);
        }
    }

    // epilogue: O / l  -> attn buffer (b, n, h*64+d) bf16
#pragma unroll
    for (int qt = 0; qt < 2; ++qt) {
        const int qrow0 = g * WIN + wave * 32 + qt * 16;
#pragma unroll
        for (int t = 0; t < 4; ++t) {
            const int col = h * HD + t * 16 + r;
#pragma unroll
            for (int j = 0; j < 8; ++j) {
                const int n = qrow0 + j + half * 8;
                const float inv = 1.0f / lrun[qt][j];
                attnb[((size_t)b * NSEQ + n) * DIM_ + col] = f2bf(O[qt][t][j] * inv);
            }
        }
    }
}

// ---------------- kernel 4: output projection (M=16384, N=1024, K=1024) ----------------
__global__ __launch_bounds__(256) void proj_gemm_kernel(const __bf16* __restrict__ A,
                                                        const __bf16* __restrict__ Wm,
                                                        const float*  __restrict__ bias,
                                                        float* __restrict__ out) {
    const int mb = blockIdx.x * 128;
    const int nb = blockIdx.y * 128;
    const int wave = threadIdx.x >> 5;
    const int wm = (wave & 3) * 32;
    const int wn = (wave >> 2) * 64;
    const int lane = threadIdx.x & 31, half = lane >> 4, r = lane & 15;

    v8f acc[2][4];
#pragma unroll
    for (int i = 0; i < 2; ++i)
#pragma unroll
        for (int j = 0; j < 4; ++j) acc[i][j] = (v8f){};

    for (int k0 = 0; k0 < DIM_; k0 += 32) {
        v16bf a0 = load_frag(A, mb + wm,      k0, DIM_);
        v16bf a1 = load_frag(A, mb + wm + 16, k0, DIM_);
        v16bf b0 = load_frag(Wm, nb + wn,      k0, DIM_);
        v16bf b1 = load_frag(Wm, nb + wn + 16, k0, DIM_);
        v16bf b2 = load_frag(Wm, nb + wn + 32, k0, DIM_);
        v16bf b3 = load_frag(Wm, nb + wn + 48, k0, DIM_);
        acc[0][0] = WMMA_BF16(a0, b0, acc[0][0]);
        acc[0][1] = WMMA_BF16(a0, b1, acc[0][1]);
        acc[0][2] = WMMA_BF16(a0, b2, acc[0][2]);
        acc[0][3] = WMMA_BF16(a0, b3, acc[0][3]);
        acc[1][0] = WMMA_BF16(a1, b0, acc[1][0]);
        acc[1][1] = WMMA_BF16(a1, b1, acc[1][1]);
        acc[1][2] = WMMA_BF16(a1, b2, acc[1][2]);
        acc[1][3] = WMMA_BF16(a1, b3, acc[1][3]);
    }

#pragma unroll
    for (int i = 0; i < 2; ++i)
#pragma unroll
        for (int j = 0; j < 4; ++j) {
            const int col = nb + wn + j * 16 + r;
            const float bv = bias[col];
#pragma unroll
            for (int jj = 0; jj < 8; ++jj) {
                const int row = mb + wm + i * 16 + jj + half * 8;
                out[(size_t)row * DIM_ + col] = acc[i][j][jj] + bv;
            }
        }
}

// ---------------- host launcher ----------------
extern "C" void kernel_launch(void* const* d_in, const int* in_sizes, int n_in,
                              void* d_out, int out_size, void* d_ws, size_t ws_size,
                              hipStream_t stream) {
    const float* x      = (const float*)d_in[0];
    const float* w_qkv  = (const float*)d_in[1];
    const float* b_qkv  = (const float*)d_in[2];
    const float* w_proj = (const float*)d_in[3];
    const float* b_proj = (const float*)d_in[4];
    float* out = (float*)d_out;

    char* ws = (char*)d_ws;
    size_t o = 0;
    const size_t M = (size_t)BATCH * NSEQ;               // 16384
    __bf16* xb    = (__bf16*)(ws + o); o += M * DIM_ * 2;
    __bf16* wqb   = (__bf16*)(ws + o); o += (size_t)3 * DIM_ * DIM_ * 2;
    __bf16* wpb   = (__bf16*)(ws + o); o += (size_t)DIM_ * DIM_ * 2;
    __bf16* Qb    = (__bf16*)(ws + o); o += M * DIM_ * 2;
    __bf16* Kb    = (__bf16*)(ws + o); o += M * DIM_ * 2;
    __bf16* Vtb   = (__bf16*)(ws + o); o += M * DIM_ * 2;
    __bf16* attnb = (__bf16*)(ws + o); o += M * DIM_ * 2;
    (void)ws_size; (void)in_sizes; (void)n_in; (void)out_size;

    cvt_bf16_kernel<<<4096, 256, 0, stream>>>(x, xb, (int)(M * DIM_));
    cvt_bf16_kernel<<<2048, 256, 0, stream>>>(w_qkv, wqb, 3 * DIM_ * DIM_);
    cvt_bf16_kernel<<<1024, 256, 0, stream>>>(w_proj, wpb, DIM_ * DIM_);

    qkv_gemm_kernel<<<dim3(M / 128, (3 * DIM_) / 128), 256, 0, stream>>>(
        xb, wqb, b_qkv, Qb, Kb, Vtb);

    attn_kernel<<<dim3(BATCH * HEADS * GRP), 256, 0, stream>>>(Qb, Kb, Vtb, attnb);

    proj_gemm_kernel<<<dim3(M / 128, DIM_ / 128), 256, 0, stream>>>(
        attnb, wpb, b_proj, out);
}
